// HierarchicalAttentionNetwork_82111184765447
// MI455X (gfx1250) — matrix operand order repaired and checked by reading, not compiled, not run on hardware
//
#include <hip/hip_runtime.h>

// ---------------------------------------------------------------------------
// HAN forward for MI455X (gfx1250): bf16 WMMA (fp32 accum) for all GEMMs,
// LDS-resident recurrent weights/state for the GRU scans (staged with
// gfx1250 async global->LDS copies), fused attention pooling, prefetch on the
// latency-critical recurrence.
// ---------------------------------------------------------------------------

#define DEV __device__ __forceinline__

typedef __attribute__((ext_vector_type(16))) __bf16 v16bf;
typedef __attribute__((ext_vector_type(8)))  float  v8f;

union FragU { v16bf v; uint4 q[2]; };

// A-matrix 16x32 bf16 fragment (row-major, leading dim ld):
// lane l -> row (l&15); K = {kb..kb+7} in vgprs 0-3, {16+kb..16+kb+7} in 4-7,
// kb = (l>=16)*8.  Two 16-byte contiguous loads.
DEV v16bf load_a_frag(const __bf16* __restrict__ base, int ld, int lane) {
  FragU f;
  const int r  = lane & 15;
  const int kb = (lane >> 4) << 3;
  f.q[0] = *(const uint4*)(base + r * ld + kb);
  f.q[1] = *(const uint4*)(base + r * ld + kb + 16);
  return f.v;
}

// B-matrix 32x16 bf16 fragment. B stored as (N x K) row-major (natural
// torch-style weight layout).  lane l -> col (l&15); K = {kb..kb+7} in
// vgprs 0-3, {16+kb..16+kb+7} in 4-7, kb = (l>=16)*8 (matches A).
DEV v16bf load_b_frag(const __bf16* __restrict__ base, int ld, int lane) {
  FragU f;
  const int nn = lane & 15;
  const int kb = (lane >> 4) << 3;
  const __bf16* p = base + nn * ld + kb;
  f.q[0] = *(const uint4*)(p);
  f.q[1] = *(const uint4*)(p + 16);
  return f.v;
}

DEV v8f wmma_bf16(v16bf a, v16bf b, v8f c) {
  return __builtin_amdgcn_wmma_f32_16x16x32_bf16(
      false, a, false, b, (short)0, c, false, false);
}

DEV float sigmoidf_(float x) { return 1.f / (1.f + __expf(-x)); }

// gfx1250 async global->LDS copy of 16 bytes per active lane (ASYNCcnt path).
DEV void async_copy_b128(unsigned ldsOff, const void* gaddr) {
  asm volatile("global_load_async_to_lds_b128 %0, %1, off"
               :: "v"(ldsOff), "v"((unsigned long long)(uintptr_t)gaddr)
               : "memory");
}
DEV void wait_asynccnt0() {
  asm volatile("s_wait_asynccnt 0x0" ::: "memory");
}

// ---------------------------------------------------------------------------
// Small utility kernels
// ---------------------------------------------------------------------------
__global__ void f32_to_bf16_k(const float* __restrict__ in,
                              __bf16* __restrict__ out, int n) {
  int i = blockIdx.x * 256 + threadIdx.x;
  if (i < n) out[i] = (__bf16)in[i];
}

__global__ void copy_f32_k(const float* __restrict__ in,
                           float* __restrict__ out, int n) {
  int i = blockIdx.x * 256 + threadIdx.x;
  if (i < n) out[i] = in[i];
}

// Embedding gather -> bf16 tokens.  PAD row (index 0) forced to zero.
__global__ void gather_emb_k(const int* __restrict__ X,
                             const float* __restrict__ emb,
                             __bf16* __restrict__ out, int nElem) {
  int i = blockIdx.x * 256 + threadIdx.x;
  if (i >= nElem) return;
  int tok = i >> 7;       // E = 128
  int e   = i & 127;
  int v   = X[tok];
  float val = (v == 0) ? 0.f : emb[v * 128 + e];
  out[i] = (__bf16)val;
}

// ---------------------------------------------------------------------------
// bf16 GEMM for N = 768:  C(Mx768, f32) = A(MxK) x B(768xK)^T + bias.
// One WG owns a 16x768 strip; each wave keeps 6 accumulators so a single
// A-fragment load feeds 6 WMMAs (6x VMEM reuse).  B (weights, <=384KB) stays
// hot in WGP$/L2; A streams exactly once.
// ---------------------------------------------------------------------------
__global__ __launch_bounds__(256) void gemm_bf16_k(
    const __bf16* __restrict__ A, const __bf16* __restrict__ Bm,
    const float* __restrict__ bias, float* __restrict__ C, int K) {
  constexpr int N = 768;
  const int wave = threadIdx.x >> 5;
  const int lane = threadIdx.x & 31;
  const int tm   = blockIdx.x;
  const __bf16* aBase = A + (size_t)tm * 16 * K;

  v8f acc[6] = {};
  for (int k0 = 0; k0 < K; k0 += 32) {
    v16bf a = load_a_frag(aBase + k0, K, lane);
#pragma unroll
    for (int j = 0; j < 6; ++j) {
      const int tn = wave + 8 * j;
      v16bf b = load_b_frag(Bm + (size_t)tn * 16 * K + k0, K, lane);
      acc[j] = wmma_bf16(a, b, acc[j]);
    }
  }
  const int m0 = tm * 16 + ((lane >> 4) << 3);
#pragma unroll
  for (int j = 0; j < 6; ++j) {
    const int n  = (wave + 8 * j) * 16 + (lane & 15);
    const float bv = bias[n];
#pragma unroll
    for (int r = 0; r < 8; ++r)
      C[(size_t)(m0 + r) * N + n] = acc[j][r] + bv;
  }
}

// ---------------------------------------------------------------------------
// GRU scan.  One WG = 16 sequences of one direction.
// LDS: w_hh (384x128 bf16, 96KB, async-staged) + h fp32 (8KB) + h bf16 mirror
//      (4KB) + gate pre-activations gh (16x384 fp32, 24KB) = 132KB / WG.
// Per step: one A-frag load feeds 3 WMMAs; next step's xp is prefetched.
// ---------------------------------------------------------------------------
__global__ __launch_bounds__(256) void gru_scan_k(
    const float* __restrict__ xp,      // (numSeq*T, 768): [dir*384 + gate*128 + c]
    const __bf16* __restrict__ whhCat, // (768, 128) bf16, rows dir*384..+383
    const float* __restrict__ bhh_f,
    const float* __restrict__ bhh_b,
    float* __restrict__ hOut,          // (numSeq*T, 256): [dir*128 + c]
    int T) {
  constexpr int OFF_HF  = 384 * 128 * 2;            // after whh
  constexpr int OFF_HBF = OFF_HF + 16 * 128 * 4;
  constexpr int OFF_GH  = OFF_HBF + 16 * 128 * 2;

  extern __shared__ char smem[];
  __bf16* whh_s = (__bf16*)smem;
  float*  h_f   = (float*)(smem + OFF_HF);
  __bf16* h_bf  = (__bf16*)(smem + OFF_HBF);
  float*  gh_s  = (float*)(smem + OFF_GH);

  const int dir = blockIdx.y;
  const int r0  = blockIdx.x * 16;
  const float* bhh = dir ? bhh_b : bhh_f;

  // Stage this direction's w_hh into LDS: 96KB = 6144 x b128 async transfers.
  {
    const unsigned ldsBase = (unsigned)(uintptr_t)whh_s;  // low 32b = LDS offset
    const char* gsrc = (const char*)(whhCat + dir * 384 * 128);
    for (int i = threadIdx.x; i < 6144; i += 256)
      async_copy_b128(ldsBase + i * 16, gsrc + i * 16);
  }
  for (int i = threadIdx.x; i < 16 * 128; i += 256) {
    h_f[i]  = 0.f;
    h_bf[i] = (__bf16)0.f;
  }
  wait_asynccnt0();
  __syncthreads();

  const int wave = threadIdx.x >> 5;
  const int lane = threadIdx.x & 31;

  for (int t = 0; t < T; ++t) {
    const int tt = dir ? (T - 1 - t) : t;

    // gh = h @ w_hh^T   (16 x 384, K = 128); wave owns tiles wave, wave+8, wave+16
    v8f acc0 = {}, acc1 = {}, acc2 = {};
#pragma unroll
    for (int k0 = 0; k0 < 128; k0 += 32) {
      v16bf a = load_a_frag(h_bf + k0, 128, lane);
      acc0 = wmma_bf16(a, load_b_frag(whh_s + (wave     ) * 2048 + k0, 128, lane), acc0);
      acc1 = wmma_bf16(a, load_b_frag(whh_s + (wave +  8) * 2048 + k0, 128, lane), acc1);
      acc2 = wmma_bf16(a, load_b_frag(whh_s + (wave + 16) * 2048 + k0, 128, lane), acc2);
    }
    {
      const int m0 = (lane >> 4) << 3;
      const int nc = lane & 15;
#pragma unroll
      for (int j = 0; j < 8; ++j) {
        gh_s[(m0 + j) * 384 + (wave      ) * 16 + nc] = acc0[j];
        gh_s[(m0 + j) * 384 + (wave +  8 ) * 16 + nc] = acc1[j];
        gh_s[(m0 + j) * 384 + (wave + 16 ) * 16 + nc] = acc2[j];
      }
    }
    __syncthreads();

    // Prefetch next timestep's xp rows (the only global operand on the
    // recurrence critical path) while this step's gates execute.
    if (t + 1 < T) {
      const int ttn  = dir ? (T - 2 - t) : t + 1;
      const int row  = threadIdx.x >> 4;
      const int chnk = threadIdx.x & 15;
      const char* p = (const char*)(xp + ((size_t)(r0 + row) * T + ttn) * 768 +
                                    dir * 384) + chnk * 96;
      __builtin_prefetch(p, 0, 0);
    }

    // Gates: r,z,n -> h update (each of 2048 elems owned by one thread).
    for (int e = threadIdx.x; e < 16 * 128; e += 256) {
      const int row = e >> 7;
      const int c   = e & 127;
      const size_t tok = (size_t)(r0 + row) * T + tt;
      const float* xrow = xp + tok * 768 + dir * 384;
      const float xr = xrow[c], xz = xrow[128 + c], xn = xrow[256 + c];
      const float hr = gh_s[row * 384 + c]       + bhh[c];
      const float hz = gh_s[row * 384 + 128 + c] + bhh[128 + c];
      const float hn = gh_s[row * 384 + 256 + c] + bhh[256 + c];
      const float r  = sigmoidf_(xr + hr);
      const float z  = sigmoidf_(xz + hz);
      const float nv = tanhf(xn + r * hn);
      const float h2 = (1.f - z) * nv + z * h_f[e];
      h_f[e]  = h2;
      h_bf[e] = (__bf16)h2;
      hOut[tok * 256 + dir * 128 + c] = h2;
    }
    __syncthreads();
  }
}

// ---------------------------------------------------------------------------
// Fused attention pool over one sequence of T timesteps, D = 256.
// score_t = ctx . tanh(h_t @ W^T + b)   (WMMA, ds_add_f32 reduction)
// a = softmax(score); out = sum_t a_t h_t
// ---------------------------------------------------------------------------
__global__ __launch_bounds__(256) void attn_pool_k(
    const float* __restrict__ h,     // (nSeq*T, 256)
    const __bf16* __restrict__ Wbf,  // (256, 256) N x K bf16
    const float* __restrict__ bias,  // (256)
    const float* __restrict__ ctx,   // (256)
    __bf16* __restrict__ outBF,      // (nSeq, 256) bf16 or nullptr
    float*  __restrict__ outF,       // (nSeq, 256) f32 or nullptr
    int T) {
  extern __shared__ char smem[];
  float*  h_f  = (float*)smem;                       // T*256 f32
  __bf16* h_bf = (__bf16*)(smem + T * 256 * 4);      // T*256 bf16
  float*  sc   = (float*)(smem + T * 256 * 4 + T * 256 * 2);  // T scores

  const int seq = blockIdx.x;
  const float* src = h + (size_t)seq * T * 256;
  for (int i = threadIdx.x; i < T * 256; i += 256) {
    const float v = src[i];
    h_f[i]  = v;
    h_bf[i] = (__bf16)v;
  }
  for (int i = threadIdx.x; i < T; i += 256) sc[i] = 0.f;
  __syncthreads();

  const int wave = threadIdx.x >> 5;
  const int lane = threadIdx.x & 31;
  // Wave owns column tiles (2*wave, 2*wave+1) for every row tile: one A-frag
  // load feeds 2 WMMAs.
  for (int tmi = 0; tmi < T / 16; ++tmi) {
    v8f acc0 = {}, acc1 = {};
    const __bf16* aB = h_bf + tmi * 16 * 256;
    const int tn0 = wave * 2, tn1 = wave * 2 + 1;
#pragma unroll
    for (int k0 = 0; k0 < 256; k0 += 32) {
      v16bf a = load_a_frag(aB + k0, 256, lane);
      acc0 = wmma_bf16(a, load_b_frag(Wbf + tn0 * 16 * 256 + k0, 256, lane), acc0);
      acc1 = wmma_bf16(a, load_b_frag(Wbf + tn1 * 16 * 256 + k0, 256, lane), acc1);
    }
    const int m0 = tmi * 16 + ((lane >> 4) << 3);
    const int n0 = tn0 * 16 + (lane & 15);
    const int n1 = tn1 * 16 + (lane & 15);
    const float cx0 = ctx[n0], bv0 = bias[n0];
    const float cx1 = ctx[n1], bv1 = bias[n1];
#pragma unroll
    for (int j = 0; j < 8; ++j) {
      atomicAdd(&sc[m0 + j], tanhf(acc0[j] + bv0) * cx0 +
                             tanhf(acc1[j] + bv1) * cx1);   // ds_add_f32
    }
  }
  __syncthreads();

  if (threadIdx.x == 0) {           // tiny softmax, T <= 64
    float mx = sc[0];
    for (int t = 1; t < T; ++t) mx = fmaxf(mx, sc[t]);
    float s = 0.f;
    for (int t = 0; t < T; ++t) { float e = __expf(sc[t] - mx); sc[t] = e; s += e; }
    const float inv = 1.f / s;
    for (int t = 0; t < T; ++t) sc[t] *= inv;
  }
  __syncthreads();

  const int d = threadIdx.x;        // D == blockDim == 256
  float acc = 0.f;
  for (int t = 0; t < T; ++t) acc += sc[t] * h_f[t * 256 + d];
  if (outBF) outBF[(size_t)seq * 256 + d] = (__bf16)acc;
  if (outF)  outF[(size_t)seq * 256 + d]  = acc;
}

// ---------------------------------------------------------------------------
// Tiny classifier: out(32x10) = doc(32x256) @ Wc(10x256)^T + bc
// ---------------------------------------------------------------------------
__global__ void classify_k(const float* __restrict__ doc,
                           const float* __restrict__ Wc,
                           const float* __restrict__ bc,
                           float* __restrict__ out) {
  int i = blockIdx.x * blockDim.x + threadIdx.x;
  if (i >= 32 * 10) return;
  const int b = i / 10, c = i % 10;
  float acc = bc[c];
  for (int k = 0; k < 256; ++k) acc += doc[b * 256 + k] * Wc[c * 256 + k];
  out[i] = acc;
}

// ---------------------------------------------------------------------------
// Host orchestration
// ---------------------------------------------------------------------------
extern "C" void kernel_launch(void* const* d_in, const int* in_sizes, int n_in,
                              void* d_out, int out_size, void* d_ws, size_t ws_size,
                              hipStream_t stream) {
  (void)in_sizes; (void)n_in; (void)out_size; (void)ws_size;
  constexpr int Bb = 32, Ss = 32, Ll = 64, Ee = 128;
  constexpr int NTOK  = Bb * Ss * Ll;   // 65536 word tokens
  constexpr int NSENT = Bb * Ss;        // 1024 sentences

  const int*   X      = (const int*)  d_in[0];
  const float* emb    = (const float*)d_in[1];
  const float* w_ih_f = (const float*)d_in[2];
  const float* w_hh_f = (const float*)d_in[3];
  const float* b_ih_f = (const float*)d_in[4];
  const float* b_hh_f = (const float*)d_in[5];
  const float* w_ih_b = (const float*)d_in[6];
  const float* w_hh_b = (const float*)d_in[7];
  const float* b_ih_b = (const float*)d_in[8];
  const float* b_hh_b = (const float*)d_in[9];
  const float* Ww     = (const float*)d_in[10];
  const float* bw     = (const float*)d_in[11];
  const float* ctx_w  = (const float*)d_in[12];
  const float* s_ih_f = (const float*)d_in[13];
  const float* s_hh_f = (const float*)d_in[14];
  const float* sb_ih_f= (const float*)d_in[15];
  const float* sb_hh_f= (const float*)d_in[16];
  const float* s_ih_b = (const float*)d_in[17];
  const float* s_hh_b = (const float*)d_in[18];
  const float* sb_ih_b= (const float*)d_in[19];
  const float* sb_hh_b= (const float*)d_in[20];
  const float* Ws     = (const float*)d_in[21];
  const float* bs     = (const float*)d_in[22];
  const float* ctx_s  = (const float*)d_in[23];
  const float* Wc     = (const float*)d_in[24];
  const float* bc     = (const float*)d_in[25];
  float* out = (float*)d_out;

  char* ws = (char*)d_ws;
  size_t off = 0;
  auto alloc = [&](size_t bytes) -> char* {
    char* p = ws + off;
    off += (bytes + 255) & ~(size_t)255;
    return p;
  };
  __bf16* wihCat  = (__bf16*)alloc(768 * 128 * 2);
  __bf16* whhCat  = (__bf16*)alloc(768 * 128 * 2);
  __bf16* sihCat  = (__bf16*)alloc(768 * 256 * 2);
  __bf16* shhCat  = (__bf16*)alloc(768 * 128 * 2);
  __bf16* WwBF    = (__bf16*)alloc(256 * 256 * 2);
  __bf16* WsBF    = (__bf16*)alloc(256 * 256 * 2);
  float*  bihCat  = (float*) alloc(768 * 4);
  float*  sbihCat = (float*) alloc(768 * 4);
  __bf16* wordsBF = (__bf16*)alloc((size_t)NTOK * Ee * 2);
  float*  xpW     = (float*) alloc((size_t)NTOK * 768 * 4);
  float*  hW      = (float*) alloc((size_t)NTOK * 256 * 4);
  __bf16* sentsBF = (__bf16*)alloc((size_t)NSENT * 256 * 2);
  float*  xpS     = (float*) alloc((size_t)NSENT * 768 * 4);
  float*  hS      = (float*) alloc((size_t)NSENT * 256 * 4);
  float*  docF    = (float*) alloc(32 * 256 * 4);

  // Allow >64KB dynamic LDS for the big kernels (132KB / 99KB).
  hipFuncSetAttribute((const void*)gru_scan_k,
                      hipFuncAttributeMaxDynamicSharedMemorySize, 160 * 1024);
  hipFuncSetAttribute((const void*)attn_pool_k,
                      hipFuncAttributeMaxDynamicSharedMemorySize, 128 * 1024);

  auto cvt = [&](const float* in, __bf16* o, int n) {
    f32_to_bf16_k<<<(n + 255) / 256, 256, 0, stream>>>(in, o, n);
  };
  cvt(w_ih_f, wihCat,               384 * 128);
  cvt(w_ih_b, wihCat + 384 * 128,   384 * 128);
  cvt(w_hh_f, whhCat,               384 * 128);
  cvt(w_hh_b, whhCat + 384 * 128,   384 * 128);
  cvt(s_ih_f, sihCat,               384 * 256);
  cvt(s_ih_b, sihCat + 384 * 256,   384 * 256);
  cvt(s_hh_f, shhCat,               384 * 128);
  cvt(s_hh_b, shhCat + 384 * 128,   384 * 128);
  cvt(Ww, WwBF, 256 * 256);
  cvt(Ws, WsBF, 256 * 256);
  copy_f32_k<<<2, 256, 0, stream>>>(b_ih_f,  bihCat,        384);
  copy_f32_k<<<2, 256, 0, stream>>>(b_ih_b,  bihCat + 384,  384);
  copy_f32_k<<<2, 256, 0, stream>>>(sb_ih_f, sbihCat,       384);
  copy_f32_k<<<2, 256, 0, stream>>>(sb_ih_b, sbihCat + 384, 384);

  // Embedding gather -> bf16
  {
    int n = NTOK * Ee;
    gather_emb_k<<<n / 256, 256, 0, stream>>>(X, emb, wordsBF, n);
  }

  // Word-level input projection: xpW = words @ [w_ih_f; w_ih_b]^T + bias
  gemm_bf16_k<<<NTOK / 16, 256, 0, stream>>>(wordsBF, wihCat, bihCat, xpW, 128);

  // Word-level GRU scan (both directions in grid.y)
  constexpr size_t SCAN_SMEM =
      384 * 128 * 2 + 16 * 128 * 4 + 16 * 128 * 2 + 16 * 384 * 4;  // 132KB
  gru_scan_k<<<dim3(NSENT / 16, 2), 256, SCAN_SMEM, stream>>>(
      xpW, whhCat, b_hh_f, b_hh_b, hW, Ll);

  // Word attention pool -> sentence vectors (bf16 for next GEMM)
  {
    size_t smem = (size_t)Ll * 256 * 4 + (size_t)Ll * 256 * 2 + Ll * 4;
    attn_pool_k<<<NSENT, 256, smem, stream>>>(hW, WwBF, bw, ctx_w,
                                              sentsBF, nullptr, Ll);
  }

  // Sentence-level input projection
  gemm_bf16_k<<<NSENT / 16, 256, 0, stream>>>(sentsBF, sihCat, sbihCat, xpS, 256);

  // Sentence-level GRU scan (32 docs x T=32)
  gru_scan_k<<<dim3(Bb / 16, 2), 256, SCAN_SMEM, stream>>>(
      xpS, shhCat, sb_hh_f, sb_hh_b, hS, Ss);

  // Doc attention pool -> doc vectors (fp32)
  {
    size_t smem = (size_t)Ss * 256 * 4 + (size_t)Ss * 256 * 2 + Ss * 4;
    attn_pool_k<<<Bb, 256, smem, stream>>>(hS, WsBF, bs, ctx_s,
                                           nullptr, docF, Ss);
  }

  // Classifier
  classify_k<<<2, 160, 0, stream>>>(docF, Wc, bc, out);
}
